// WaveGlow_60507499266372
// MI455X (gfx1250) — matrix operand order repaired and checked by reading, not compiled, not run on hardware
//
#include <hip/hip_runtime.h>

#define HD 256          // hidden dim
#define LN 8            // layers
#define TM 256          // samples per workgroup
#define NSLOPE 0.01f    // leaky relu slope

typedef __attribute__((ext_vector_type(16))) _Float16 v16h;
typedef __attribute__((ext_vector_type(8)))  _Float16 v8h;
typedef __attribute__((ext_vector_type(2)))  _Float16 v2h;
typedef __attribute__((ext_vector_type(8)))  float    v8f;

// ---------------------------------------------------------------------------
// Pre-pass: pack sW2/tW2 (L,256,256 f32, row-major [k][n]) into f16 WMMA
// B-fragment layout for V_WMMA_F32_16X16X32_F16 (wave32):
//   fragment id: (((l*2+net)*8 + k)*16 + nt)   -> 32 lanes x 16 halves
//   lane L: column n = nt*16 + (L&15)
//   half h (0..15): K = k*32 + (L>>4)*16 + h   (pairs packed per dword)
// ---------------------------------------------------------------------------
__global__ void pack_w2_kernel(const float* __restrict__ sW2,
                               const float* __restrict__ tW2,
                               _Float16* __restrict__ pack)
{
  const int b  = blockIdx.x;        // (l*2+net)*8 + k   : 128 blocks
  const int k  = b & 7;
  const int ln = b >> 3;
  const float* W = (ln & 1) ? tW2 : sW2;
  const int l  = ln >> 1;
  for (int i = threadIdx.x; i < 8192; i += 256) {
    const int h    = i & 15;
    const int lane = (i >> 4) & 31;
    const int nt   = i >> 9;
    const int K = k * 32 + (lane >> 4) * 16 + h;
    const int n = nt * 16 + (lane & 15);
    pack[(size_t)b * 8192 + i] = (_Float16)W[(l * HD + K) * HD + n];
  }
}

// ---------------------------------------------------------------------------
// Fused flow kernel: 256 samples per block, all 8 coupling layers in one pass.
// ---------------------------------------------------------------------------
__global__ __launch_bounds__(256)
void flow_fused_kernel(const float* __restrict__ z,
                       const float* __restrict__ mask,
                       const float* __restrict__ sW1, const float* __restrict__ sb1,
                       const float* __restrict__ sb2,
                       const float* __restrict__ sW3, const float* __restrict__ sb3,
                       const float* __restrict__ tW1, const float* __restrict__ tb1,
                       const float* __restrict__ tb2,
                       const float* __restrict__ tW3, const float* __restrict__ tb3,
                       const float* __restrict__ Winv,
                       const _Float16* __restrict__ pack,
                       float* __restrict__ out)
{
  // dynamic LDS: h1 activations (TM x HD f16 = 128KB) + staged B frags (128KB)
  extern __shared__ _Float16 dynsmem[];
  _Float16* __restrict__ h1  = dynsmem;             // TM*HD halves
  _Float16* __restrict__ Bsm = dynsmem + TM * HD;   // 8*16*512 halves

  __shared__ float xbuf[TM * 2];
  __shared__ float souts[TM], touts[TM];
  __shared__ float w1s[HD], b1s[HD], b2s[HD], w3s[HD];
  __shared__ float Wc[4];
  __shared__ int   iaS;

  const int tid   = threadIdx.x;
  const int wv    = tid >> 5;
  const int lane  = tid & 31;
  const int lrow  = lane & 15;
  const int lhalf = lane >> 4;

  // load 256 samples (512 floats) coalesced
  ((float2*)xbuf)[tid] = ((const float2*)z)[(size_t)blockIdx.x * 256 + tid];
  __syncthreads();

  for (int l = 0; l < LN; ++l) {
    // --- per-layer setup: W = inv(Winv[l]) (2x2), active dim from mask ---
    if (tid == 0) {
      const float a = Winv[l*4+0], b = Winv[l*4+1];
      const float c = Winv[l*4+2], d = Winv[l*4+3];
      const float inv = 1.0f / (a * d - b * c);
      Wc[0] =  d * inv; Wc[1] = -b * inv;
      Wc[2] = -c * inv; Wc[3] =  a * inv;
      iaS = (mask[l * 2] > 0.5f) ? 0 : 1;
    }
    __syncthreads();

    // --- invertible 1x1 conv: x = x @ W ---
    {
      const float x0 = xbuf[tid*2+0], x1 = xbuf[tid*2+1];
      xbuf[tid*2+0] = x0 * Wc[0] + x1 * Wc[2];
      xbuf[tid*2+1] = x0 * Wc[1] + x1 * Wc[3];
    }
    const int ia = iaS, io = 1 - iaS;
    __syncthreads();

    for (int net = 0; net < 2; ++net) {
      const float* __restrict__ W1 = net ? tW1 : sW1;
      const float* __restrict__ B1 = net ? tb1 : sb1;
      const float* __restrict__ B2 = net ? tb2 : sb2;
      const float* __restrict__ W3 = net ? tW3 : sW3;

      // preload per-net constants into LDS (active W1 row, b1, b2, W3 column io)
      w1s[tid] = W1[(l * 2 + ia) * HD + tid];
      b1s[tid] = B1[l * HD + tid];
      b2s[tid] = B2[l * HD + tid];
      w3s[tid] = W3[(l * HD + tid) * 2 + io];
      __syncthreads();

      // --- fill h1 = leaky(x_active * w1 + b1) as f16, and stage B frags ---
      {
        const int jp = tid & 127;           // column pair (constant per thread)
        const int mb = tid >> 7;            // 0/1 : row parity
        const float wa = w1s[jp*2+0], wb = w1s[jp*2+1];
        const float ba = b1s[jp*2+0], bb = b1s[jp*2+1];
        #pragma unroll 4
        for (int it = 0; it < 128; ++it) {
          const int m = it * 2 + mb;
          const float xa = xbuf[m * 2 + ia];
          float v0 = xa * wa + ba;
          float v1 = xa * wb + bb;
          v0 = v0 > 0.f ? v0 : v0 * NSLOPE;
          v1 = v1 > 0.f ? v1 : v1 * NSLOPE;
          v2h p; p[0] = (_Float16)v0; p[1] = (_Float16)v1;
          *(v2h*)&h1[m * HD + jp * 2] = p;
        }
        // stage 64KB of packed W2 fragments (shared by all 8 waves)
        const v8h* __restrict__ src =
            (const v8h*)(pack + (size_t)(l * 2 + net) * 65536);
        v8h* __restrict__ dst = (v8h*)Bsm;
        #pragma unroll
        for (int it = 0; it < 16; ++it)
          dst[it * 256 + tid] = src[it * 256 + tid];
      }
      __syncthreads();

      // --- WMMA GEMM: h2 = leaky(h1 @ W2 + b2); fused dot with w3 column ---
      // A fragments (16x32 f16): lane row = mt*16+lrow;
      //   halves: K = k*32 + lhalf*16 + {0..15}
      v16h A[2][8];
      #pragma unroll
      for (int mtl = 0; mtl < 2; ++mtl) {
        const int m = (wv * 2 + mtl) * 16 + lrow;
        #pragma unroll
        for (int k = 0; k < 8; ++k) {
          const v8h a0 = *(const v8h*)&h1[m * HD + k * 32 + lhalf * 16];
          const v8h a1 = *(const v8h*)&h1[m * HD + k * 32 + lhalf * 16 + 8];
          #pragma unroll
          for (int i = 0; i < 8; ++i) {
            A[mtl][k][i]     = a0[i];
            A[mtl][k][8 + i] = a1[i];
          }
        }
      }

      float acc0[8], acc1[8];
      #pragma unroll
      for (int r = 0; r < 8; ++r) { acc0[r] = 0.f; acc1[r] = 0.f; }

      const _Float16* __restrict__ Bbase = Bsm + lane * 16;

      for (int nt = 0; nt < 16; ++nt) {
        // epilogue scalars early, share the wait window with the B loads
        const int ng = nt * 16 + lrow;
        const float bias = b2s[ng];
        const float w3v  = w3s[ng];

        v8f c0 = {}; v8f c1 = {};
        v16h Bf[4];

        // group 1: k = 0..3 -> one dscnt wait for 8 loads, then 8 WMMAs
        #pragma unroll
        for (int k = 0; k < 4; ++k)
          Bf[k] = *(const v16h*)&Bbase[(k * 16 + nt) * 512];
        #pragma unroll
        for (int k = 0; k < 4; ++k) {
          c0 = __builtin_amdgcn_wmma_f32_16x16x32_f16(
                   false, A[0][k], false, Bf[k], (short)0, c0, false, false);
          c1 = __builtin_amdgcn_wmma_f32_16x16x32_f16(
                   false, A[1][k], false, Bf[k], (short)0, c1, false, false);
        }
        // group 2: k = 4..7 (loads issue while group-1 WMMAs execute)
        #pragma unroll
        for (int k = 4; k < 8; ++k)
          Bf[k - 4] = *(const v16h*)&Bbase[(k * 16 + nt) * 512];
        #pragma unroll
        for (int k = 4; k < 8; ++k) {
          c0 = __builtin_amdgcn_wmma_f32_16x16x32_f16(
                   false, A[0][k], false, Bf[k - 4], (short)0, c0, false, false);
          c1 = __builtin_amdgcn_wmma_f32_16x16x32_f16(
                   false, A[1][k], false, Bf[k - 4], (short)0, c1, false, false);
        }

        // epilogue: bias + leaky + multiply by w3 column, accumulate
        #pragma unroll
        for (int r = 0; r < 8; ++r) {
          float h = c0[r] + bias; h = h > 0.f ? h : h * NSLOPE; acc0[r] += h * w3v;
          float g = c1[r] + bias; g = g > 0.f ? g : g * NSLOPE; acc1[r] += g * w3v;
        }
      }

      // butterfly reduce across the 16 N-lanes of each half-group
      #pragma unroll
      for (int o = 1; o < 16; o <<= 1) {
        #pragma unroll
        for (int r = 0; r < 8; ++r) {
          acc0[r] += __shfl_xor(acc0[r], o, 32);
          acc1[r] += __shfl_xor(acc1[r], o, 32);
        }
      }
      float* __restrict__ ov = net ? touts : souts;
      if (lrow == 0) {
        #pragma unroll
        for (int r = 0; r < 8; ++r) {
          ov[(wv * 2 + 0) * 16 + lhalf * 8 + r] = acc0[r];
          ov[(wv * 2 + 1) * 16 + lhalf * 8 + r] = acc1[r];
        }
      }
      __syncthreads();
    } // net

    // --- coupling update: x[io] = x[io]*exp(tanh(s)) + t ---
    {
      const float sraw = souts[tid] + sb3[l * 2 + io];
      const float traw = touts[tid] + tb3[l * 2 + io];
      const float sv = tanhf(sraw);
      const float xi = xbuf[tid * 2 + io];
      xbuf[tid * 2 + io] = xi * __expf(sv) + traw;
    }
    __syncthreads();
  } // layers

  ((float2*)out)[(size_t)blockIdx.x * 256 + tid] = ((float2*)xbuf)[tid];
}

// ---------------------------------------------------------------------------
extern "C" void kernel_launch(void* const* d_in, const int* in_sizes, int n_in,
                              void* d_out, int out_size, void* d_ws, size_t ws_size,
                              hipStream_t stream) {
  const float* z    = (const float*)d_in[0];
  const float* mask = (const float*)d_in[1];
  const float* sW1  = (const float*)d_in[2];
  const float* sb1  = (const float*)d_in[3];
  const float* sW2  = (const float*)d_in[4];
  const float* sb2  = (const float*)d_in[5];
  const float* sW3  = (const float*)d_in[6];
  const float* sb3  = (const float*)d_in[7];
  const float* tW1  = (const float*)d_in[8];
  const float* tb1  = (const float*)d_in[9];
  const float* tW2  = (const float*)d_in[10];
  const float* tb2  = (const float*)d_in[11];
  const float* tW3  = (const float*)d_in[12];
  const float* tb3  = (const float*)d_in[13];
  const float* Winv = (const float*)d_in[14];

  _Float16* pack = (_Float16*)d_ws;   // 2 MB: f16 WMMA-fragment-packed W2 weights

  pack_w2_kernel<<<128, 256, 0, stream>>>(sW2, tW2, pack);

  const int nSamples = in_sizes[0] / 2;     // B
  const int nBlocks  = nSamples / TM;       // 4096
  const size_t dynLds = (size_t)2 * TM * HD * sizeof(_Float16);  // 256 KB

  flow_fused_kernel<<<nBlocks, 256, dynLds, stream>>>(
      z, mask, sW1, sb1, sb2, sW3, sb3,
      tW1, tb1, tb2, tW3, tb3, Winv, pack, (float*)d_out);
}
